// LSTM_32615981646347
// MI455X (gfx1250) — matrix operand order repaired and checked by reading, not compiled, not run on hardware
//
#include <hip/hip_runtime.h>
#include <hip/hip_bf16.h>

typedef __attribute__((ext_vector_type(16))) __bf16 v16bf;
typedef __attribute__((ext_vector_type(8)))  float  v8f;

#define Bsz   32
#define Tsz   2048
#define Dsz   256
#define Hsz   256
#define G4H   1024          // 4*H
#define KT    8             // K tiles of 32 over D=256
#define KTL   4             // K tiles of Wh cached in LDS
#define CT    64            // column tiles of 16 over 4H=1024
#define PACKN (KT*CT*32*8)  // 131072 uint32 per packed matrix
#define WHL_N (KTL*CT*32*8) // 65536 uint32 (256 KB) LDS-cached Wh

union FragBF { unsigned int u[8]; uint4 u4[2]; v16bf v; };
union FragF  { v8f v; float f[8]; };

// ---- precise bf16 (RNE) for the one-time weight pack -----------------------
static __device__ __forceinline__ unsigned short f2bf_rne(float f) {
    unsigned int u = __float_as_uint(f);
    u = (u + 0x7FFFu + ((u >> 16) & 1u)) >> 16;
    return (unsigned short)u;
}
static __device__ __forceinline__ unsigned int pack2_rne(float lo, float hi) {
    return (unsigned int)f2bf_rne(lo) | ((unsigned int)f2bf_rne(hi) << 16);
}
// ---- 1-instruction bf16x2 pack (truncate) for hot loops: v_perm_b32 --------
static __device__ __forceinline__ unsigned int pack2_fast(float lo, float hi) {
    return __builtin_amdgcn_perm(__float_as_uint(hi), __float_as_uint(lo), 0x07060302u);
}
// ---- fast activations (v_exp_f32 / v_rcp_f32) ------------------------------
static __device__ __forceinline__ float sigm(float x) {
    return __builtin_amdgcn_rcpf(1.0f + __expf(-x));
}
static __device__ __forceinline__ float ftanh(float x) {
    x = fminf(fmaxf(x, -15.0f), 15.0f);
    float t = __expf(2.0f * x);
    return (t - 1.0f) * __builtin_amdgcn_rcpf(t + 1.0f);
}

// ---------------------------------------------------------------------------
// Kernel 1: pack Wi (gid<PACKN) and Wh (gid>=PACKN) f32[256][1024] into the
// CDNA5 WMMA B-operand layout:  P[kt][ct][lane][j] = bf16x2(W[k],W[k+1]) at
// column ct*16+(lane&15), k = kt*32 + (j>=4?16:0) + (lane>=16?8:0) + 2(j&3).
// ---------------------------------------------------------------------------
__global__ __launch_bounds__(256) void lstm_pack_kernel(
    const float* __restrict__ Wi, const float* __restrict__ Wh,
    unsigned int* __restrict__ WiP, unsigned int* __restrict__ WhP) {
    unsigned int gid = blockIdx.x * 256u + threadIdx.x;
    if (gid >= 2u * PACKN) return;
    const float* W = (gid < PACKN) ? Wi : Wh;
    unsigned int* P = (gid < PACKN) ? WiP : WhP;
    unsigned int w    = gid & (PACKN - 1u);
    unsigned int j    =  w        & 7u;
    unsigned int lane = (w >> 3)  & 31u;
    unsigned int ct   = (w >> 8)  & 63u;
    unsigned int kt   = (w >> 14) & 7u;
    unsigned int k = kt * 32u + ((j & 4u) ? 16u : 0u) + ((lane >= 16u) ? 8u : 0u) + 2u * (j & 3u);
    unsigned int n = ct * 16u + (lane & 15u);
    P[w] = pack2_rne(W[k * G4H + n], W[(k + 1u) * G4H + n]);
}

// ---------------------------------------------------------------------------
// Kernel 2: Zx[b*T+t][c] = X[b,t,:] @ Wi[:,c] + bias[c]   (bf16 WMMA, f32 acc)
// One wave per 16-row tile x 4 column tiles. 65536 waves total.
// ---------------------------------------------------------------------------
__global__ __launch_bounds__(256) void lstm_gemm_kernel(
    const float* __restrict__ X, const unsigned int* __restrict__ WiP,
    const float* __restrict__ bias, float* __restrict__ Zx) {
    const unsigned int wave = blockIdx.x * 8u + (threadIdx.x >> 5);
    const unsigned int lane = threadIdx.x & 31u;
    const unsigned int rt = wave >> 4;       // row tile (0..4095)
    const unsigned int cg = wave & 15u;      // column group (4 tiles each)
    const unsigned int M  = lane & 15u;
    const unsigned int hi = lane >> 4;

    v8f acc[4];
    v8f z8 = {0.f,0.f,0.f,0.f,0.f,0.f,0.f,0.f};
    #pragma unroll
    for (int c = 0; c < 4; ++c) acc[c] = z8;

    const float* xrow = X + (size_t)(rt * 16u + M) * Dsz;

    #pragma unroll
    for (int kt = 0; kt < KT; ++kt) {
        // slots 0..3: K = kt*32 + hi*8 + {0..7};  slots 4..7: +16
        FragBF a;
        const float4* xlo = (const float4*)(xrow + kt * 32 + hi * 8);
        const float4* xhi = (const float4*)(xrow + kt * 32 + 16 + hi * 8);
        float4 x0 = xlo[0], x1 = xlo[1], x2 = xhi[0], x3 = xhi[1];
        a.u[0] = pack2_fast(x0.x, x0.y);  a.u[1] = pack2_fast(x0.z, x0.w);
        a.u[2] = pack2_fast(x1.x, x1.y);  a.u[3] = pack2_fast(x1.z, x1.w);
        a.u[4] = pack2_fast(x2.x, x2.y);  a.u[5] = pack2_fast(x2.z, x2.w);
        a.u[6] = pack2_fast(x3.x, x3.y);  a.u[7] = pack2_fast(x3.z, x3.w);
        #pragma unroll
        for (int c = 0; c < 4; ++c) {
            unsigned int ctg = cg * 4u + c;
            FragBF bm;
            const uint4* bp = (const uint4*)(WiP + (((kt * CT + ctg) * 32u + lane) * 8u));
            bm.u4[0] = bp[0];
            bm.u4[1] = bp[1];
            acc[c] = __builtin_amdgcn_wmma_f32_16x16x32_bf16(
                false, a.v, false, bm.v, (short)0, acc[c], false, false);
        }
    }
    #pragma unroll
    for (int c = 0; c < 4; ++c) {
        unsigned int ctg = cg * 4u + c;
        unsigned int col = ctg * 16u + (lane & 15u);
        float bv = bias[col];
        FragF cf; cf.v = acc[c];
        #pragma unroll
        for (int r = 0; r < 8; ++r) {
            unsigned int row = rt * 16u + (unsigned)r + hi * 8u;
            Zx[(size_t)row * G4H + col] = cf.f[r] + bv;
        }
    }
}

// ---------------------------------------------------------------------------
// Kernel 3: persistent recurrence. 2 blocks (batch halves) x 512 threads.
// Wave w owns output columns [16w,16w+16) for ALL FOUR gates (column tiles
// {w, w+16, w+32, w+48}), so gates are fused in the WMMA accumulators and
// c-state lives in VGPRs. h is double-buffered as bf16 in LDS (one barrier
// per step). Wh k-tiles 0..3 are served from LDS (DS pipe), 4..7 from L2
// (VMEM pipe) so both operand streams run concurrently.
// ---------------------------------------------------------------------------
__global__ __launch_bounds__(512) void lstm_rec_kernel(
    const unsigned int* __restrict__ WhP, const float* __restrict__ Zx,
    const int* __restrict__ lengths, float* __restrict__ out) {
    __shared__ unsigned short hbs[2][16 * Hsz];   // 2 x 8 KB bf16 h (double buffer)
    __shared__ unsigned int   whl[WHL_N];         // 256 KB Wh k-tiles 0..3

    const unsigned int tid  = threadIdx.x;
    const unsigned int wave = tid >> 5;
    const unsigned int lane = tid & 31u;
    const unsigned int half = blockIdx.x;         // 0 or 1
    const unsigned int M    = lane & 15u;         // A-row / C-column-local
    const unsigned int hi   = lane >> 4;

    // cooperative fill: Wh k-tiles 0..3 -> LDS (coalesced b128)
    {
        const uint4* src = (const uint4*)WhP;
        uint4* dst = (uint4*)whl;
        for (unsigned int i = tid; i < WHL_N / 4u; i += 512u) dst[i] = src[i];
    }
    for (unsigned int i = tid; i < 16u * Hsz; i += 512u) hbs[0][i] = 0;

    // per-lane persistent state: c for rows m = r + hi*8, column wave*16+M
    float creg[8];
    int tcapv[8];
    const float* zrow[8];
    #pragma unroll
    for (int r = 0; r < 8; ++r) {
        creg[r] = 0.0f;
        unsigned int m = (unsigned)r + hi * 8u;
        unsigned int b = half * 16u + m;
        int tl = lengths[b] - 1; if (tl < 0) tl = 0;
        tcapv[r] = tl;
        zrow[r] = Zx + (size_t)b * Tsz * G4H + wave * 16u + M;
    }
    __syncthreads();

    for (int t = 0; t < Tsz; ++t) {
        const unsigned int rb = (unsigned)t & 1u, wb = rb ^ 1u;

        // ---- prefetch this step's Zx gate slices (hidden behind WMMAs)
        float zxv[4][8];
        #pragma unroll
        for (int r = 0; r < 8; ++r) {
            const float* zp = zrow[r] + (size_t)t * G4H;
            #pragma unroll
            for (int g = 0; g < 4; ++g) zxv[g][r] = zp[g * 256];
        }

        // ---- z_h = h @ Wh: gate g accumulates column tile g*16 + wave
        FragF acc[4];
        v8f z8 = {0.f,0.f,0.f,0.f,0.f,0.f,0.f,0.f};
        #pragma unroll
        for (int g = 0; g < 4; ++g) acc[g].v = z8;

        #pragma unroll
        for (int kt = 0; kt < KT; ++kt) {
            FragBF a;   // two ds_load_b128 from the read h-buffer
            a.u4[0] = *(const uint4*)&hbs[rb][M * Hsz + (unsigned)kt * 32u + hi * 8u];
            a.u4[1] = *(const uint4*)&hbs[rb][M * Hsz + (unsigned)kt * 32u + 16u + hi * 8u];
            #pragma unroll
            for (int g = 0; g < 4; ++g) {
                unsigned int ctg = (unsigned)g * 16u + wave;
                unsigned int base = ((unsigned)kt * CT + ctg) * 32u * 8u + lane * 8u;
                FragBF bm;
                if (kt < KTL) {                       // LDS-resident half (DS pipe)
                    bm.u4[0] = *(const uint4*)&whl[base];
                    bm.u4[1] = *(const uint4*)&whl[base + 4u];
                } else {                              // L2-resident half (VMEM pipe)
                    const uint4* bp = (const uint4*)(WhP + base);
                    bm.u4[0] = bp[0];
                    bm.u4[1] = bp[1];
                }
                acc[g].v = __builtin_amdgcn_wmma_f32_16x16x32_bf16(
                    false, a.v, false, bm.v, (short)0, acc[g].v, false, false);
            }
        }

        // ---- fused gates in registers + state update + output capture
        #pragma unroll
        for (int r = 0; r < 8; ++r) {
            unsigned int m = (unsigned)r + hi * 8u;
            float zi = acc[0].f[r] + zxv[0][r];
            float zf = acc[1].f[r] + zxv[1][r];
            float zg = acc[2].f[r] + zxv[2][r];
            float zo = acc[3].f[r] + zxv[3][r];
            float cnew = sigm(zf) * creg[r] + sigm(zi) * ftanh(zg);
            creg[r] = cnew;
            float hnew = sigm(zo) * ftanh(cnew);
            hbs[wb][m * Hsz + wave * 16u + M] =
                (unsigned short)(__float_as_uint(hnew) >> 16);   // ds_store_b16
            if (t == tcapv[r])
                out[(half * 16u + m) * Hsz + wave * 16u + M] = hnew;
        }
        __syncthreads();   // single barrier: h(t+1) visible, buffers swap
    }
}

extern "C" void kernel_launch(void* const* d_in, const int* in_sizes, int n_in,
                              void* d_out, int out_size, void* d_ws, size_t ws_size,
                              hipStream_t stream) {
    const float* X   = (const float*)d_in[0];   // [32,2048,256]
    const int* lens  = (const int*)  d_in[1];   // [32]
    const float* Wi  = (const float*)d_in[2];   // [256,1024]
    const float* Wh  = (const float*)d_in[3];   // [256,1024]
    const float* bia = (const float*)d_in[4];   // [1024]
    float* out       = (float*)d_out;           // [32,256]

    unsigned int* WiP = (unsigned int*)d_ws;                 // 512 KB
    unsigned int* WhP = WiP + PACKN;                         // 512 KB
    float* Zx = (float*)((char*)d_ws + (1u << 20));          // 256 MB

    lstm_pack_kernel<<<(2 * PACKN + 255) / 256, 256, 0, stream>>>(Wi, Wh, WiP, WhP);
    // 4096 row tiles * 16 col groups = 65536 waves, 8 waves/block
    lstm_gemm_kernel<<<8192, 256, 0, stream>>>(X, WiP, bia, Zx);
    lstm_rec_kernel<<<2, 512, 0, stream>>>(WhP, Zx, lens, out);
}
